// GNODE_17222818857331
// MI455X (gfx1250) — matrix operand order repaired and checked by reading, not compiled, not run on hardware
//
#include <hip/hip_runtime.h>
#include <hip/hip_bf16.h>

typedef __attribute__((ext_vector_type(16))) _Float16 v16h;
typedef __attribute__((ext_vector_type(8)))  float    v8f;

#define N_NODES 100000
#define N_EDGES 1600000
#define LATENT  10
#define ODE_D   32
#define N_STEPS 10

// output layout: dec_n [0,1e5) | dec_e [1e5,1.7e6) | next_pos [1.7e6,1.8e6) | next_vel [1.8e6,1.9e6)
#define OUT_DECN 0
#define OUT_DECE 100000
#define OUT_POS  1700000
#define OUT_VEL  1800000

__device__ __forceinline__ void atomic_add_f32(float* p, float v) {
  __hip_atomic_fetch_add(p, v, __ATOMIC_RELAXED, __HIP_MEMORY_SCOPE_AGENT);
}

// single-instruction ReLU: v_med3_f32(x, 0, +inf) clamps to [0, inf)
__device__ __forceinline__ float relu1(float x) {
  return __builtin_amdgcn_fmed3f(x, 0.0f, __builtin_inff());
}

// ---------------- zero sent/recv accumulators ----------------
__global__ void k_zero(float* p, int n) {
  int i = blockIdx.x * blockDim.x + threadIdx.x;
  if (i < n) p[i] = 0.0f;
}

// ---------------- edges: encode + decode + scatter ----------------
__global__ void k_edges(const float* __restrict__ edges,
                        const int* __restrict__ snd, const int* __restrict__ rcv,
                        const float* __restrict__ W, const float* __restrict__ b,
                        const float* __restrict__ dW, const float* __restrict__ db,
                        float* __restrict__ sent, float* __restrict__ recvb,
                        float* __restrict__ dec_e) {
  int e = blockIdx.x * blockDim.x + threadIdx.x;
  if (e >= N_EDGES) return;
  float e0 = edges[e * 3 + 0], e1 = edges[e * 3 + 1], e2 = edges[e * 3 + 2];
  int s = snd[e], r = rcv[e];
  float he[LATENT];
  float dec = db[0];
#pragma unroll
  for (int j = 0; j < LATENT; ++j) {
    float h = b[j] + e0 * W[j] + e1 * W[10 + j] + e2 * W[20 + j];
    he[j] = h;
    dec += h * dW[j];
  }
  dec_e[e] = dec;
#pragma unroll
  for (int j = 0; j < LATENT; ++j) {
    atomic_add_f32(&sent[s * LATENT + j], he[j]);
    atomic_add_f32(&recvb[r * LATENT + j], he[j]);
  }
}

// ---------------- nodes: encode + assemble x (100000 x 32) ----------------
__global__ void k_build_x(const float* __restrict__ nodes,
                          const float* __restrict__ sent, const float* __restrict__ recvb,
                          const float* __restrict__ g,
                          const float* __restrict__ W, const float* __restrict__ b,
                          float* __restrict__ x) {
  int n = blockIdx.x * blockDim.x + threadIdx.x;
  if (n >= N_NODES) return;
  float n0 = nodes[2 * n], n1 = nodes[2 * n + 1];
  float* xr = x + n * ODE_D;
#pragma unroll
  for (int j = 0; j < LATENT; ++j) xr[j] = b[j] + n0 * W[j] + n1 * W[10 + j];
#pragma unroll
  for (int j = 0; j < LATENT; ++j) xr[10 + j] = sent[n * LATENT + j];
#pragma unroll
  for (int j = 0; j < LATENT; ++j) xr[20 + j] = recvb[n * LATENT + j];
  xr[30] = g[0];
  xr[31] = g[1];
}

// ---------------- ODE tile: 16 nodes x 32 features ----------------
// D layout (f32): lane = node; lanes<16 hold feature-slots {r, 16+r}, lanes>=16 hold {8+r, 24+r}.
// B layout (f16, AMD table order): lanes<16 hold K=0..7,16..23, lanes>=16 hold K=8..15,24..31.
// These K equal the D slots, and A rows use the same order (row = (u&7)+8*hi+16*(u>=8)),
// so D -> next B is a pure within-lane f32->f16 convert (v_cvt_pk_f16_f32): no cross-lane traffic.
struct T16 { v8f t0; v8f t1; };   // t0 = slots (hi?8:0)+r, t1 = slots 16+(hi?8:0)+r

__device__ __forceinline__ v16h make_B(const v8f z0, const v8f z1) {
  v16h bb;
#pragma unroll
  for (int r = 0; r < 8; ++r) {
    bb[r]     = (_Float16)z0[r];
    bb[8 + r] = (_Float16)z1[r];
  }
  return bb;
}

// deriv(y) = relu(y@W1+b1)@W2+b2, computed transposed: D = (W^T)(y^T) + bias
__device__ __forceinline__ T16 deriv(const T16 y,
                                     const v16h A1t0, const v16h A1t1,
                                     const v16h A2t0, const v16h A2t1,
                                     const v8f b1t0, const v8f b1t1,
                                     const v8f b2t0, const v8f b2t1) {
  v16h B = make_B(y.t0, y.t1);
  v8f h0 = __builtin_amdgcn_wmma_f32_16x16x32_f16(false, A1t0, false, B, (short)0, b1t0, false, false);
  v8f h1 = __builtin_amdgcn_wmma_f32_16x16x32_f16(false, A1t1, false, B, (short)0, b1t1, false, false);
#pragma unroll
  for (int i = 0; i < 8; ++i) { h0[i] = relu1(h0[i]); h1[i] = relu1(h1[i]); }
  v16h B2 = make_B(h0, h1);
  T16 o;
  o.t0 = __builtin_amdgcn_wmma_f32_16x16x32_f16(false, A2t0, false, B2, (short)0, b2t0, false, false);
  o.t1 = __builtin_amdgcn_wmma_f32_16x16x32_f16(false, A2t1, false, B2, (short)0, b2t1, false, false);
  return o;
}

__global__ __launch_bounds__(256) void k_ode(
    const float* __restrict__ x,
    const float* __restrict__ W1, const float* __restrict__ b1,
    const float* __restrict__ W2, const float* __restrict__ b2,
    const float* __restrict__ nodeW, const float* __restrict__ nodeB,
    const float* __restrict__ decW, const float* __restrict__ decB,
    const float* __restrict__ nodes, float* __restrict__ out) {
  const int lane = threadIdx.x & 31;
  const int wave = threadIdx.x >> 5;
  const int m = lane & 15;
  const bool hi = lane >= 16;
  const int nodeRaw = blockIdx.x * 128 + wave * 16 + m;
  const int node = (nodeRaw < N_NODES) ? nodeRaw : (N_NODES - 1);

  // ---- W^T tiles as WMMA A operands (16x32 f16) in AMD's table K order ----
  // A element index u = 2v+q  (VGPR v, halfword q); logical W row = (u&7) + 8*hi + (u>=8 ? 16 : 0).
  auto loadA = [&](const float* W, int colBase) {
    v16h a;
#pragma unroll
    for (int u = 0; u < 16; ++u) {
      int row = (u & 7) + (hi ? 8 : 0) + (u >= 8 ? 16 : 0);
      a[u] = (_Float16)W[row * ODE_D + colBase + m];
    }
    return a;
  };
  const v16h A1t0 = loadA(W1, 0), A1t1 = loadA(W1, 16);
  const v16h A2t0 = loadA(W2, 0), A2t1 = loadA(W2, 16);

  // ---- biases in C/D layout: VGPR r -> slot (base + r + hi*8) ----
  auto loadBias = [&](const float* b, int base) {
    v8f c;
#pragma unroll
    for (int r = 0; r < 8; ++r) c[r] = b[base + r + (hi ? 8 : 0)];
    return c;
  };
  const v8f b1t0 = loadBias(b1, 0), b1t1 = loadBias(b1, 16);
  const v8f b2t0 = loadBias(b2, 0), b2t1 = loadBias(b2, 16);

  // ---- load y0 = x[node] into D layout (contiguous float4 groups) ----
  T16 y;
  {
    const float4* xr = (const float4*)(x + (long)node * ODE_D);
    int o = hi ? 2 : 0;
    float4 q0 = xr[o], q1 = xr[o + 1], q2 = xr[o + 4], q3 = xr[o + 5];
    y.t0[0]=q0.x; y.t0[1]=q0.y; y.t0[2]=q0.z; y.t0[3]=q0.w;
    y.t0[4]=q1.x; y.t0[5]=q1.y; y.t0[6]=q1.z; y.t0[7]=q1.w;
    y.t1[0]=q2.x; y.t1[1]=q2.y; y.t1[2]=q2.z; y.t1[3]=q2.w;
    y.t1[4]=q3.x; y.t1[5]=q3.y; y.t1[6]=q3.z; y.t1[7]=q3.w;
  }

  // ---- Dormand-Prince RK45, 10 steps, y lives entirely in registers ----
  const float H = 0.1f;
  const float hA21 = H * (1.f/5.f);
  const float hA31 = H * (3.f/40.f),       hA32 = H * (9.f/40.f);
  const float hA41 = H * (44.f/45.f),      hA42 = H * (-56.f/15.f),      hA43 = H * (32.f/9.f);
  const float hA51 = H * (19372.f/6561.f), hA52 = H * (-25360.f/2187.f), hA53 = H * (64448.f/6561.f), hA54 = H * (-212.f/729.f);
  const float hA61 = H * (9017.f/3168.f),  hA62 = H * (-355.f/33.f),     hA63 = H * (46732.f/5247.f), hA64 = H * (49.f/176.f), hA65 = H * (-5103.f/18656.f);
  const float hB1 = H * (35.f/384.f), hB3 = H * (500.f/1113.f), hB4 = H * (125.f/192.f), hB5 = H * (-2187.f/6784.f), hB6 = H * (11.f/84.f);

  for (int step = 0; step < N_STEPS; ++step) {
    T16 k1 = deriv(y, A1t0, A1t1, A2t0, A2t1, b1t0, b1t1, b2t0, b2t1);
    T16 yt;
    yt.t0 = y.t0 + k1.t0 * hA21;
    yt.t1 = y.t1 + k1.t1 * hA21;
    T16 k2 = deriv(yt, A1t0, A1t1, A2t0, A2t1, b1t0, b1t1, b2t0, b2t1);
    yt.t0 = y.t0 + k1.t0 * hA31 + k2.t0 * hA32;
    yt.t1 = y.t1 + k1.t1 * hA31 + k2.t1 * hA32;
    T16 k3 = deriv(yt, A1t0, A1t1, A2t0, A2t1, b1t0, b1t1, b2t0, b2t1);
    yt.t0 = y.t0 + k1.t0 * hA41 + k2.t0 * hA42 + k3.t0 * hA43;
    yt.t1 = y.t1 + k1.t1 * hA41 + k2.t1 * hA42 + k3.t1 * hA43;
    T16 k4 = deriv(yt, A1t0, A1t1, A2t0, A2t1, b1t0, b1t1, b2t0, b2t1);
    yt.t0 = y.t0 + k1.t0 * hA51 + k2.t0 * hA52 + k3.t0 * hA53 + k4.t0 * hA54;
    yt.t1 = y.t1 + k1.t1 * hA51 + k2.t1 * hA52 + k3.t1 * hA53 + k4.t1 * hA54;
    T16 k5 = deriv(yt, A1t0, A1t1, A2t0, A2t1, b1t0, b1t1, b2t0, b2t1);
    yt.t0 = y.t0 + k1.t0 * hA61 + k2.t0 * hA62 + k3.t0 * hA63 + k4.t0 * hA64 + k5.t0 * hA65;
    yt.t1 = y.t1 + k1.t1 * hA61 + k2.t1 * hA62 + k3.t1 * hA63 + k4.t1 * hA64 + k5.t1 * hA65;
    T16 k6 = deriv(yt, A1t0, A1t1, A2t0, A2t1, b1t0, b1t1, b2t0, b2t1);
    y.t0 = y.t0 + k1.t0 * hB1 + k3.t0 * hB3 + k4.t0 * hB4 + k5.t0 * hB5 + k6.t0 * hB6;
    y.t1 = y.t1 + k1.t1 * hB1 + k3.t1 * hB3 + k4.t1 * hB4 + k5.t1 * hB5 + k6.t1 * hB6;
  }

  // ---- decode head folded to one 32-vector: w[f] = sum_j node_out_W[f][j]*dec_node_W[j] ----
  float acc = 0.0f;
#pragma unroll
  for (int r = 0; r < 8; ++r) {
    int f0 = r + (hi ? 8 : 0);
    int f1 = 16 + f0;
    float w0 = 0.0f, w1 = 0.0f;
#pragma unroll
    for (int j = 0; j < LATENT; ++j) {
      float dwj = decW[j];
      w0 += nodeW[f0 * LATENT + j] * dwj;
      w1 += nodeW[f1 * LATENT + j] * dwj;
    }
    acc += y.t0[r] * w0 + y.t1[r] * w1;
  }
  acc += __shfl_xor(acc, 16, 32);   // combine both feature halves of this node

  if (!hi && nodeRaw < N_NODES) {
    float c = decB[0];
#pragma unroll
    for (int j = 0; j < LATENT; ++j) c += nodeB[j] * decW[j];
    float dec = acc + c;
    float vel = nodes[2 * nodeRaw + 1] + dec;   // DT = 1
    out[OUT_DECN + nodeRaw] = dec;
    out[OUT_VEL + nodeRaw] = vel;
    out[OUT_POS + nodeRaw] = nodes[2 * nodeRaw] + vel;
  }
}

extern "C" void kernel_launch(void* const* d_in, const int* in_sizes, int n_in,
                              void* d_out, int out_size, void* d_ws, size_t ws_size,
                              hipStream_t stream) {
  const float* nodes      = (const float*)d_in[0];
  const float* edges      = (const float*)d_in[1];
  const int*   senders    = (const int*)  d_in[2];
  const int*   receivers  = (const int*)  d_in[3];
  const float* globals_   = (const float*)d_in[4];
  const float* enc_node_W = (const float*)d_in[5];
  const float* enc_node_b = (const float*)d_in[6];
  const float* enc_edge_W = (const float*)d_in[7];
  const float* enc_edge_b = (const float*)d_in[8];
  const float* ode_W1     = (const float*)d_in[9];
  const float* ode_b1     = (const float*)d_in[10];
  const float* ode_W2     = (const float*)d_in[11];
  const float* ode_b2     = (const float*)d_in[12];
  const float* node_out_W = (const float*)d_in[13];
  const float* node_out_b = (const float*)d_in[14];
  const float* dec_node_W = (const float*)d_in[15];
  const float* dec_node_b = (const float*)d_in[16];
  const float* dec_edge_W = (const float*)d_in[17];
  const float* dec_edge_b = (const float*)d_in[18];
  float* out = (float*)d_out;

  float* sent  = (float*)d_ws;                 // 100000*10 f32
  float* recvb = sent + N_NODES * LATENT;      // 100000*10 f32
  float* x     = recvb + N_NODES * LATENT;     // 100000*32 f32

  // 1) zero scatter accumulators (must happen every call)
  {
    int n = 2 * N_NODES * LATENT;
    k_zero<<<(n + 255) / 256, 256, 0, stream>>>(sent, n);
  }
  // 2) edge encode + decode + segment-sum scatter (L2-resident atomics)
  k_edges<<<(N_EDGES + 255) / 256, 256, 0, stream>>>(
      edges, senders, receivers, enc_edge_W, enc_edge_b, dec_edge_W, dec_edge_b,
      sent, recvb, out + OUT_DECE);
  // 3) node encode + assemble ODE state x
  k_build_x<<<(N_NODES + 255) / 256, 256, 0, stream>>>(
      nodes, sent, recvb, globals_, enc_node_W, enc_node_b, x);
  // 4) fused RK45 ODE (WMMA) + decode + integrate pos/vel
  k_ode<<<(N_NODES + 127) / 128, 256, 0, stream>>>(
      x, ode_W1, ode_b1, ode_W2, ode_b2,
      node_out_W, node_out_b, dec_node_W, dec_node_b, nodes, out);
}
